// ModulatedRichDecoder_32573031973089
// MI455X (gfx1250) — matrix operand order repaired and checked by reading, not compiled, not run on hardware
//
#include <hip/hip_runtime.h>
#include <hip/hip_bf16.h>
#include <math.h>

typedef __attribute__((ext_vector_type(16))) _Float16 v16h;
typedef __attribute__((ext_vector_type(8)))  _Float16 v8h;
typedef __attribute__((ext_vector_type(8)))  float    v8f;

union F16x16 { v16h v; v8h h[2]; };

#define NB 16      // batch
#define TT 512     // time
#define DD 256     // model dim
#define NHD 4      // heads
#define DH 64      // head dim
#define FF 1024    // ff dim
#define HH 512     // gru hidden
#define EE 64      // energy dim
#define OO 317     // output dim
#define LL 2
#define BT (NB*TT) // 8192
#define OHW ((long)OO*HH) // 162304

__device__ __forceinline__ float geluf(float x) {
    return 0.5f * x * (1.0f + tanhf(0.7978845608028654f * (x + 0.044715f * x * x * x)));
}
__device__ __forceinline__ float sigf(float x) { return 1.0f / (1.0f + expf(-x)); }

// ---------------------------------------------------------------------------
// Generic batched WMMA GEMM: C[z] = act( scale * A[z](MxK) * Bt[z](NxK)^T + bias )
// A,Bt are f16 row-major (Bt is B transposed: Bt[n][k] = B[k][n]).
// One wave computes one 16x16 C tile; 8 waves per block along N.
// z decomposed as (zb = z/nh, zh = z%nh) with independent strides.
// ---------------------------------------------------------------------------
__global__ __launch_bounds__(256) void k_gemm(
    const _Float16* __restrict__ A, const _Float16* __restrict__ Bt,
    float* __restrict__ Cf, _Float16* __restrict__ Ch,
    const float* __restrict__ bias, int biasZ,
    int M, int N, int K, int lda, int ldb, int ldc, int nh,
    long sAb, long sAh, long sBb, long sBh, long sCb, long sCh,
    float scale, int act)
{
    int wave = threadIdx.x >> 5;
    int lane = threadIdx.x & 31;
    int tn = blockIdx.x * 8 + wave;
    int n0 = tn * 16;
    if (n0 >= N) return;                      // whole-wave uniform exit (EXEC stays full)
    int m0 = blockIdx.y * 16;
    int z  = blockIdx.z;
    int zb = z / nh, zh = z % nh;

    const _Float16* a_base = A  + zb * sAb + zh * sAh;
    const _Float16* b_base = Bt + zb * sBb + zh * sBh;

    int g   = lane >> 4;
    int l15 = lane & 15;
    int arow  = m0 + l15;
    int bcol  = n0 + l15;
    int bcolc = bcol < N ? bcol : N - 1;      // clamp loads for ragged N

    const _Float16* pa = a_base + (long)arow  * lda + g * 16;
    const _Float16* pb = b_base + (long)bcolc * ldb + g * 16;

    v8f acc = {};
    for (int k = 0; k < K; k += 32) {
        F16x16 a, b;
        a.h[0] = *(const v8h*)(pa + k);       // K = 16g .. 16g+7
        a.h[1] = *(const v8h*)(pa + k + 8);   // K = 16g+8 .. 16g+15
        b.h[0] = *(const v8h*)(pb + k);       // K = 16g .. 16g+7
        b.h[1] = *(const v8h*)(pb + k + 8);   // K = 16g+8 .. 16g+15
        acc = __builtin_amdgcn_wmma_f32_16x16x32_f16(
            false, a.v, false, b.v, (short)0, acc, false, false);
    }

    float bv = 0.0f;
    if (bias) bv = bias[(long)z * biasZ + bcolc];
    if (bcol < N) {
        long cofs = zb * sCb + zh * sCh;
        #pragma unroll
        for (int i = 0; i < 8; ++i) {
            int row = m0 + i + 8 * g;
            float v = acc[i] * scale + bv;
            if (act == 1) v = geluf(v);
            long addr = cofs + (long)row * ldc + bcol;
            if (Ch) Ch[addr] = (_Float16)v;
            else    Cf[addr] = v;
        }
    }
}

// ---------------------------------------------------------------------------
// Weight prep: f32 -> f16 transpose, f32 -> f16 copy
// ---------------------------------------------------------------------------
__global__ void k_trans16(const float* __restrict__ src, _Float16* __restrict__ dst,
                          int R, int C) {
    long i = (long)blockIdx.x * 256 + threadIdx.x;
    if (i >= (long)R * C) return;
    int  c = (int)(i % C);
    long r = i / C;
    dst[(long)c * R + r] = (_Float16)src[i];
}
__global__ void k_copy16(const float* __restrict__ src, _Float16* __restrict__ dst, long n) {
    long i = (long)blockIdx.x * 256 + threadIdx.x;
    if (i < n) dst[i] = (_Float16)src[i];
}

// ---------------------------------------------------------------------------
// Hypernet: effW[b,o,h] = head_w[o,h] + hyp_bW[o*H+h] + sum_e es[b,e]*hyp_wW[e,o*H+h]
// ---------------------------------------------------------------------------
__global__ void k_hypw(const float* __restrict__ es, const float* __restrict__ wW,
                       const float* __restrict__ bW, const float* __restrict__ headw,
                       _Float16* __restrict__ effW) {
    __shared__ float esl[NB * EE];
    int tid = threadIdx.x;
    for (int i = tid; i < NB * EE; i += 256) esl[i] = es[i];
    __syncthreads();
    long idx = (long)blockIdx.x * 256 + tid;
    if (idx >= OHW) return;
    float base = headw[idx] + bW[idx];
    float acc[NB];
    #pragma unroll
    for (int b = 0; b < NB; ++b) acc[b] = base;
    for (int e = 0; e < EE; ++e) {
        float w = wW[(long)e * OHW + idx];
        #pragma unroll
        for (int b = 0; b < NB; ++b) acc[b] += esl[b * EE + e] * w;
    }
    #pragma unroll
    for (int b = 0; b < NB; ++b) effW[(long)b * OHW + idx] = (_Float16)acc[b];
}

__global__ void k_hypb(const float* __restrict__ es, const float* __restrict__ wb,
                       const float* __restrict__ bb, const float* __restrict__ headb,
                       float* __restrict__ effb) {
    int i = blockIdx.x * 256 + threadIdx.x;
    if (i >= NB * OO) return;
    int b = i / OO, o = i % OO;
    float acc = headb[o] + bb[o];
    for (int e = 0; e < EE; ++e) acc += es[b * EE + e] * wb[e * OO + o];
    effb[i] = acc;
}

// ---------------------------------------------------------------------------
// Prenet: per (b,t): h = gelu([f0,loud] @ w1 + b1); x = h @ w2 + b2
// ---------------------------------------------------------------------------
__global__ void k_prenet(const float* __restrict__ f0, const float* __restrict__ loud,
                         const float* __restrict__ w1, const float* __restrict__ b1,
                         const float* __restrict__ w2, const float* __restrict__ b2,
                         float* __restrict__ x0, _Float16* __restrict__ x16) {
    __shared__ float h[DD];
    long row = blockIdx.x;
    int d = threadIdx.x;
    float a = f0[row], l = loud[row];
    h[d] = geluf(a * w1[d] + l * w1[DD + d] + b1[d]);
    __syncthreads();
    float acc = b2[d];
    for (int k = 0; k < DD; ++k) acc += h[k] * w2[(long)k * DD + d];
    x0[row * DD + d]  = acc;
    x16[row * DD + d] = (_Float16)acc;
}

// ---------------------------------------------------------------------------
// LayerNorm of (xin + res) over D=256 (one block per row); emits f32 + f16
// ---------------------------------------------------------------------------
__global__ void k_ln(const float* __restrict__ xin, const float* __restrict__ res,
                     const float* __restrict__ gg, const float* __restrict__ bb,
                     float* __restrict__ xout, _Float16* __restrict__ x16) {
    __shared__ float red[DD];
    long row = blockIdx.x;
    int d = threadIdx.x;
    float v = xin[row * DD + d] + res[row * DD + d];
    red[d] = v; __syncthreads();
    for (int s = 128; s > 0; s >>= 1) { if (d < s) red[d] += red[d + s]; __syncthreads(); }
    float m = red[0] * (1.0f / DD); __syncthreads();
    float c = v - m;
    red[d] = c * c; __syncthreads();
    for (int s = 128; s > 0; s >>= 1) { if (d < s) red[d] += red[d + s]; __syncthreads(); }
    float var = red[0] * (1.0f / DD);
    float o = c * rsqrtf(var + 1e-5f) * gg[d] + bb[d];
    xout[row * DD + d] = o;
    x16[row * DD + d]  = (_Float16)o;
}

// ---------------------------------------------------------------------------
// Causal softmax over rows of length T=512 (scores already scaled); writes f16 P
// ---------------------------------------------------------------------------
__global__ void k_softmax(const float* __restrict__ S, _Float16* __restrict__ P) {
    __shared__ float red[256];
    long row = blockIdx.x;         // over B*NH*T
    int  t   = (int)(row & (TT - 1));
    const float* s = S + row * TT;
    _Float16*    p = P + row * TT;
    int tid = threadIdx.x;
    float v0 = (tid       <= t) ? s[tid]       : -INFINITY;
    float v1 = (tid + 256 <= t) ? s[tid + 256] : -INFINITY;
    red[tid] = fmaxf(v0, v1); __syncthreads();
    for (int s2 = 128; s2 > 0; s2 >>= 1) { if (tid < s2) red[tid] = fmaxf(red[tid], red[tid + s2]); __syncthreads(); }
    float mx = red[0]; __syncthreads();
    float e0 = (tid       <= t) ? expf(v0 - mx) : 0.0f;
    float e1 = (tid + 256 <= t) ? expf(v1 - mx) : 0.0f;
    red[tid] = e0 + e1; __syncthreads();
    for (int s2 = 128; s2 > 0; s2 >>= 1) { if (tid < s2) red[tid] += red[tid + s2]; __syncthreads(); }
    float inv = 1.0f / red[0];
    p[tid]       = (_Float16)(e0 * inv);
    p[tid + 256] = (_Float16)(e1 * inv);
}

// ---------------------------------------------------------------------------
// V transpose per batch: (B,T,D) -> (B,D,T), f16
// ---------------------------------------------------------------------------
__global__ void k_vtrans(const _Float16* __restrict__ v, _Float16* __restrict__ vt) {
    long i = (long)blockIdx.x * 256 + threadIdx.x;
    if (i >= (long)NB * TT * DD) return;
    int  d = (int)(i & (DD - 1));
    long r = i >> 8;               // /DD
    int  t = (int)(r & (TT - 1));
    long b = r >> 9;               // /TT
    vt[(b * DD + d) * TT + t] = v[i];
}

// ---------------------------------------------------------------------------
// Persistent single-workgroup GRU. 1024 threads = 32 waves.
// LDS: gh f32[16*1536] | h32 f32[16*512] | h16 f16[16*512]  (= 147456 B)
// Per step: gh = h16 @ wh^T + bh via WMMA (96 N-tiles, 3/wave), then gates.
// ---------------------------------------------------------------------------
#define GRU_LDS (NB*3*HH*4 + NB*HH*4 + NB*HH*2)
__global__ __launch_bounds__(1024) void k_gru(
    const float* __restrict__ gx,      // (B,T,3H) with bi
    const _Float16* __restrict__ wh,   // (3H,H) f16 (Bt form)
    const float* __restrict__ bh,      // (3H)
    float* __restrict__ hs)            // (B,T,H)
{
    extern __shared__ char smem[];
    float*    ghs = (float*)smem;                                   // 16*1536
    float*    h32 = (float*)(smem + NB * 3 * HH * 4);               // 16*512
    _Float16* h16 = (_Float16*)(smem + NB * 3 * HH * 4 + NB * HH * 4);

    int tid  = threadIdx.x;
    int wave = tid >> 5, lane = tid & 31;
    int g = lane >> 4, l15 = lane & 15;

    for (int i = tid; i < NB * HH; i += 1024) { h32[i] = 0.0f; h16[i] = (_Float16)0.0f; }
    __syncthreads();

    for (int t = 0; t < TT; ++t) {
        // gh = h @ wh^T + bh ; M=16(batch), N=1536, K=512
        for (int tt = 0; tt < 3; ++tt) {
            int n0   = (wave + tt * 32) * 16;
            int bcol = n0 + l15;
            const _Float16* pb = wh  + (long)bcol * HH + g * 16;
            const _Float16* pa = h16 + l15 * HH + g * 16;
            v8f acc = {};
            for (int k = 0; k < HH; k += 32) {
                F16x16 a, b;
                a.h[0] = *(const v8h*)(pa + k);
                a.h[1] = *(const v8h*)(pa + k + 8);
                b.h[0] = *(const v8h*)(pb + k);
                b.h[1] = *(const v8h*)(pb + k + 8);
                acc = __builtin_amdgcn_wmma_f32_16x16x32_f16(
                    false, a.v, false, b.v, (short)0, acc, false, false);
            }
            float bb = bh[bcol];
            #pragma unroll
            for (int i = 0; i < 8; ++i)
                ghs[(i + 8 * g) * (3 * HH) + bcol] = acc[i] + bb;
        }
        __syncthreads();

        // gates: each thread handles 8 consecutive (b,j)
        {
            int e0 = tid * 8;
            int b  = e0 >> 9;          // /512
            int j  = e0 & (HH - 1);
            const float* gxp = gx + ((long)(b * TT + t)) * (3 * HH);
            #pragma unroll
            for (int u = 0; u < 8; ++u) {
                int jj = j + u;
                float xr = gxp[jj], xz = gxp[HH + jj], xn = gxp[2 * HH + jj];
                float hr = ghs[b * 3 * HH + jj];
                float hz = ghs[b * 3 * HH + HH + jj];
                float hn = ghs[b * 3 * HH + 2 * HH + jj];
                float r  = sigf(xr + hr);
                float zz = sigf(xz + hz);
                float nn = tanhf(xn + r * hn);
                float hnew = (1.0f - zz) * nn + zz * h32[b * HH + jj];
                h32[b * HH + jj] = hnew;
                h16[b * HH + jj] = (_Float16)hnew;
                hs[((long)b * TT + t) * HH + jj] = hnew;
            }
        }
        __syncthreads();
    }
}

// ---------------------------------------------------------------------------
// Head nonlinearities + output scatter into concatenated tuple layout
// ---------------------------------------------------------------------------
__global__ void k_heads(const float* __restrict__ raw, float* __restrict__ out) {
    long i = (long)blockIdx.x * 256 + threadIdx.x;
    if (i >= (long)BT * OO) return;
    int  o  = (int)(i % OO);
    long bt = i / OO;
    float v = raw[i];
    float s = sigf(v);
    const long O1 = (long)BT * 128, O2 = O1 + BT, O3 = O2 + (long)BT * 16,
               O4 = O3 + (long)BT * 8, O5 = O4 + (long)BT * 4, O6 = O5 + (long)BT * 128;
    if      (o < 128) out[bt * 128 + o] = s;
    else if (o < 129) out[O1 + bt] = tanhf(v) * 0.5f;
    else if (o < 145) out[O2 + bt * 16 + (o - 129)] = s;
    else if (o < 153) out[O3 + bt * 8  + (o - 145)] = s;
    else if (o < 157) out[O4 + bt * 4  + (o - 153)] = s;
    else if (o < 285) out[O5 + bt * 128 + (o - 157)] = s;
    else              out[O6 + bt * 32  + (o - 285)] = s;
}

// ===========================================================================
extern "C" void kernel_launch(void* const* d_in, const int* in_sizes, int n_in,
                              void* d_out, int out_size, void* d_ws, size_t ws_size,
                              hipStream_t stream) {
    const float* f0    = (const float*)d_in[0];
    const float* loud  = (const float*)d_in[1];
    const float* es    = (const float*)d_in[2];
    const float* pw1   = (const float*)d_in[3];
    const float* pb1   = (const float*)d_in[4];
    const float* pw2   = (const float*)d_in[5];
    const float* pb2   = (const float*)d_in[6];
    const float* wq    = (const float*)d_in[7];
    const float* bq    = (const float*)d_in[8];
    const float* wk    = (const float*)d_in[9];
    const float* bk    = (const float*)d_in[10];
    const float* wv    = (const float*)d_in[11];
    const float* bv    = (const float*)d_in[12];
    const float* wo    = (const float*)d_in[13];
    const float* bo    = (const float*)d_in[14];
    const float* ln1g  = (const float*)d_in[15];
    const float* ln1b  = (const float*)d_in[16];
    const float* fw1   = (const float*)d_in[17];
    const float* fb1   = (const float*)d_in[18];
    const float* fw2   = (const float*)d_in[19];
    const float* fb2   = (const float*)d_in[20];
    const float* ln2g  = (const float*)d_in[21];
    const float* ln2b  = (const float*)d_in[22];
    const float* projw = (const float*)d_in[23];
    const float* projb = (const float*)d_in[24];
    const float* gwi   = (const float*)d_in[25];
    const float* gwh   = (const float*)d_in[26];
    const float* gbi   = (const float*)d_in[27];
    const float* gbh   = (const float*)d_in[28];
    const float* postw = (const float*)d_in[29];
    const float* postb = (const float*)d_in[30];
    const float* headw = (const float*)d_in[31];
    const float* headb = (const float*)d_in[32];
    const float* hypwW = (const float*)d_in[33];
    const float* hypbW = (const float*)d_in[34];
    const float* hypwb = (const float*)d_in[35];
    const float* hypbb = (const float*)d_in[36];
    float* outp = (float*)d_out;

    // ---- workspace layout ----
    char* w = (char*)d_ws;
    auto alloc = [&](size_t bytes) -> char* {
        char* p = w; w += (bytes + 255) & ~(size_t)255; return p;
    };
    _Float16* WQT   = (_Float16*)alloc((size_t)LL * DD * DD * 2);
    _Float16* WKT   = (_Float16*)alloc((size_t)LL * DD * DD * 2);
    _Float16* WVT   = (_Float16*)alloc((size_t)LL * DD * DD * 2);
    _Float16* WOT   = (_Float16*)alloc((size_t)LL * DD * DD * 2);
    _Float16* FF1T  = (_Float16*)alloc((size_t)LL * FF * DD * 2);
    _Float16* FF2T  = (_Float16*)alloc((size_t)LL * DD * FF * 2);
    _Float16* PROJT = (_Float16*)alloc((size_t)HH * DD * 2);
    _Float16* WI16  = (_Float16*)alloc((size_t)3 * HH * HH * 2);
    _Float16* WH16  = (_Float16*)alloc((size_t)3 * HH * HH * 2);
    _Float16* POSTT = (_Float16*)alloc((size_t)HH * HH * 2);
    _Float16* EFFW  = (_Float16*)alloc((size_t)NB * OHW * 2);
    float*    EFFB  = (float*)alloc((size_t)NB * OO * 4);
    float*    X0    = (float*)alloc((size_t)BT * DD * 4);
    float*    Y0    = (float*)alloc((size_t)BT * DD * 4);
    _Float16* X16   = (_Float16*)alloc((size_t)BT * DD * 2);
    _Float16* Q16   = (_Float16*)alloc((size_t)BT * DD * 2);
    _Float16* K16   = (_Float16*)alloc((size_t)BT * DD * 2);
    _Float16* V16   = (_Float16*)alloc((size_t)BT * DD * 2);
    _Float16* VT16  = (_Float16*)alloc((size_t)BT * DD * 2);
    _Float16* AO16  = (_Float16*)alloc((size_t)BT * DD * 2);
    // region A: scores (attn) / gx (gru) share
    char* regA = alloc((size_t)NB * NHD * TT * TT * 4);          // 67 MB
    float* SCORES = (float*)regA;
    float* GX     = (float*)regA;
    // region B: P16 (attn) / ffh16 (ff) / hs16+post16+raw (tail) share
    char* regB = alloc((size_t)NB * NHD * TT * TT * 2);          // 33.5 MB
    _Float16* P16    = (_Float16*)regB;
    _Float16* FFH16  = (_Float16*)regB;
    _Float16* HS16   = (_Float16*)regB;
    _Float16* POST16 = (_Float16*)(regB + (size_t)BT * HH * 2);
    float*    RAW    = (float*)(regB + (size_t)BT * HH * 4);
    float*    HS     = (float*)alloc((size_t)BT * HH * 4);
    _Float16* XP16   = (_Float16*)alloc((size_t)BT * HH * 2);

    auto blocks = [](long n) { return (unsigned)((n + 255) / 256); };
    auto gemm = [&](const _Float16* A, const _Float16* B, float* Cf, _Float16* Ch,
                    const float* bias, int biasZ,
                    int M, int N, int Kd, int lda, int ldb, int ldc,
                    int Z, int nh, long sAb, long sAh, long sBb, long sBh,
                    long sCb, long sCh, float scale, int act) {
        dim3 grid((unsigned)(((N + 15) / 16 + 7) / 8), (unsigned)(M / 16), (unsigned)Z);
        k_gemm<<<grid, 256, 0, stream>>>(A, B, Cf, Ch, bias, biasZ, M, N, Kd,
                                         lda, ldb, ldc, nh, sAb, sAh, sBb, sBh,
                                         sCb, sCh, scale, act);
    };

    // ---- weight prep (f32 -> f16, transpose to Bt) ----
    for (int l = 0; l < LL; ++l) {
        k_trans16<<<blocks((long)DD * DD), 256, 0, stream>>>(wq + (long)l * DD * DD, WQT + (long)l * DD * DD, DD, DD);
        k_trans16<<<blocks((long)DD * DD), 256, 0, stream>>>(wk + (long)l * DD * DD, WKT + (long)l * DD * DD, DD, DD);
        k_trans16<<<blocks((long)DD * DD), 256, 0, stream>>>(wv + (long)l * DD * DD, WVT + (long)l * DD * DD, DD, DD);
        k_trans16<<<blocks((long)DD * DD), 256, 0, stream>>>(wo + (long)l * DD * DD, WOT + (long)l * DD * DD, DD, DD);
        k_trans16<<<blocks((long)DD * FF), 256, 0, stream>>>(fw1 + (long)l * DD * FF, FF1T + (long)l * FF * DD, DD, FF);
        k_trans16<<<blocks((long)FF * DD), 256, 0, stream>>>(fw2 + (long)l * FF * DD, FF2T + (long)l * DD * FF, FF, DD);
    }
    k_trans16<<<blocks((long)DD * HH), 256, 0, stream>>>(projw, PROJT, DD, HH);
    k_copy16<<<blocks((long)3 * HH * HH), 256, 0, stream>>>(gwi, WI16, (long)3 * HH * HH);
    k_copy16<<<blocks((long)3 * HH * HH), 256, 0, stream>>>(gwh, WH16, (long)3 * HH * HH);
    k_trans16<<<blocks((long)HH * HH), 256, 0, stream>>>(postw, POSTT, HH, HH);

    // ---- hypernet (eff_W, eff_b) ----
    k_hypw<<<blocks(OHW), 256, 0, stream>>>(es, hypwW, hypbW, headw, EFFW);
    k_hypb<<<blocks((long)NB * OO), 256, 0, stream>>>(es, hypwb, hypbb, headb, EFFB);

    // ---- prenet ----
    k_prenet<<<BT, DD, 0, stream>>>(f0, loud, pw1, pb1, pw2, pb2, X0, X16);

    // ---- transformer layers ----
    for (int l = 0; l < LL; ++l) {
        long ldd = (long)l * DD * DD;
        gemm(X16, WQT + ldd, nullptr, Q16, bq + l * DD, 0, BT, DD, DD, DD, DD, DD,
             1, 1, 0, 0, 0, 0, 0, 0, 1.0f, 0);
        gemm(X16, WKT + ldd, nullptr, K16, bk + l * DD, 0, BT, DD, DD, DD, DD, DD,
             1, 1, 0, 0, 0, 0, 0, 0, 1.0f, 0);
        gemm(X16, WVT + ldd, nullptr, V16, bv + l * DD, 0, BT, DD, DD, DD, DD, DD,
             1, 1, 0, 0, 0, 0, 0, 0, 1.0f, 0);
        k_vtrans<<<blocks((long)BT * DD), 256, 0, stream>>>(V16, VT16);
        // scores[b,h] = (Q K^T)/8
        gemm(Q16, K16, SCORES, nullptr, nullptr, 0, TT, TT, DH, DD, DD, TT,
             NB * NHD, NHD, (long)TT * DD, DH, (long)TT * DD, DH,
             (long)NHD * TT * TT, (long)TT * TT, 0.125f, 0);
        k_softmax<<<NB * NHD * TT, 256, 0, stream>>>(SCORES, P16);
        // attn out = P @ V  (per head, written into its column slice)
        gemm(P16, VT16, nullptr, AO16, nullptr, 0, TT, DH, TT, TT, TT, DD,
             NB * NHD, NHD, (long)NHD * TT * TT, (long)TT * TT,
             (long)DD * TT, (long)DH * TT, (long)TT * DD, DH, 1.0f, 0);
        gemm(AO16, WOT + ldd, Y0, nullptr, bo + l * DD, 0, BT, DD, DD, DD, DD, DD,
             1, 1, 0, 0, 0, 0, 0, 0, 1.0f, 0);
        k_ln<<<BT, DD, 0, stream>>>(X0, Y0, ln1g + l * DD, ln1b + l * DD, X0, X16);
        gemm(X16, FF1T + (long)l * FF * DD, nullptr, FFH16, fb1 + l * FF, 0,
             BT, FF, DD, DD, DD, FF, 1, 1, 0, 0, 0, 0, 0, 0, 1.0f, 1);
        gemm(FFH16, FF2T + (long)l * DD * FF, Y0, nullptr, fb2 + l * DD, 0,
             BT, DD, FF, FF, FF, DD, 1, 1, 0, 0, 0, 0, 0, 0, 1.0f, 0);
        k_ln<<<BT, DD, 0, stream>>>(X0, Y0, ln2g + l * DD, ln2b + l * DD, X0, X16);
    }

    // ---- projection D -> H ----
    gemm(X16, PROJT, nullptr, XP16, projb, 0, BT, HH, DD, DD, DD, HH,
         1, 1, 0, 0, 0, 0, 0, 0, 1.0f, 0);
    // ---- gx = xp @ wi^T + bi ----
    gemm(XP16, WI16, GX, nullptr, gbi, 0, BT, 3 * HH, HH, HH, HH, 3 * HH,
         1, 1, 0, 0, 0, 0, 0, 0, 1.0f, 0);
    // ---- GRU scan (single persistent workgroup, WMMA recurrence) ----
    k_gru<<<1, 1024, GRU_LDS, stream>>>(GX, WH16, gbh, HS);
    k_copy16<<<blocks((long)BT * HH), 256, 0, stream>>>(HS, HS16, (long)BT * HH);
    // ---- post: gelu(hs @ post_w + b) ----
    gemm(HS16, POSTT, nullptr, POST16, postb, 0, BT, HH, HH, HH, HH, HH,
         1, 1, 0, 0, 0, 0, 0, 0, 1.0f, 1);
    // ---- head: raw[b] = post[b] @ effW[b]^T + effb[b] ----
    gemm(POST16, EFFW, RAW, nullptr, EFFB, OO, TT, OO, HH, HH, HH, OO,
         NB, 1, (long)TT * HH, 0, OHW, 0, (long)TT * OO, 0, 1.0f, 0);
    // ---- output nonlinearities + scatter ----
    k_heads<<<blocks((long)BT * OO), 256, 0, stream>>>(RAW, outp);
}
